// VanillaRNN_91216515432752
// MI455X (gfx1250) — matrix operand order repaired and checked by reading, not compiled, not run on hardware
//
#include <hip/hip_runtime.h>

typedef __attribute__((ext_vector_type(16))) _Float16 v16h;
typedef __attribute__((ext_vector_type(8)))  float    v8f;

#define BATCH   2048
#define SEQ     1024
#define HID     128
#define CLASSES 10
#define ROWS    16          // batch rows per workgroup tile
#define HSTRIDE 136         // LDS row stride in halves: 272B rows, 16B aligned, padded
#define NWAVES  4           // waves per workgroup (split over N)
#define NBW     2           // N-blocks (16 cols each) per wave: 4*2*16 = 128 = HID

#if __has_builtin(__builtin_amdgcn_tanhf)
__device__ __forceinline__ float fast_tanh(float v) { return __builtin_amdgcn_tanhf(v); }
#else
__device__ __forceinline__ float fast_tanh(float v) { return tanhf(v); }
#endif

// 4 waves cooperate on one 16-row batch tile. h lives in a double-buffered LDS
// tile (f16); each wave owns 2 of the 8 16-wide N-blocks of h_new per step and
// runs them on v_wmma_f32_16x16x32_f16 with W_hh held in registers. One
// workgroup barrier per timestep publishes h to the other waves.
__global__ __launch_bounds__(NWAVES * 32) void rnn_wmma_kernel(
    const float* __restrict__ x,       // [BATCH, SEQ]
    const float* __restrict__ W_hx,    // [1, HID]
    const float* __restrict__ W_hh,    // [HID, HID]
    const float* __restrict__ W_ph,    // [HID, CLASSES]
    const float* __restrict__ bias_h,  // [HID]
    const float* __restrict__ bias_p,  // [CLASSES]
    float* __restrict__ out)           // [BATCH, CLASSES]
{
    __shared__ _Float16 hbuf[2][ROWS * HSTRIDE];

    const int tid   = threadIdx.x;
    const int wave  = tid >> 5;      // 0..3
    const int lane  = tid & 31;      // wave32 lane
    const int col16 = lane & 15;
    const int hi    = lane >> 4;     // half-wave selector
    const int r0    = blockIdx.x * ROWS;

    // ---- h0 = 0 (only buffer 0 is read at t=0) ----
    for (int i = tid; i < ROWS * HSTRIDE; i += NWAVES * 32)
        hbuf[0][i] = (_Float16)0.0f;

    // ---- this wave's W_hh B-operands (f16), per (K-chunk c, local N-block b) ----
    // B 32x16 layout: lanes 0-15 hold K=0..15, lanes 16-31 hold K=16..31;
    // N = (wave*NBW + b)*16 + (lane&15).
    v16h b_hh[4][NBW];
    #pragma unroll
    for (int c = 0; c < 4; ++c) {
        #pragma unroll
        for (int b = 0; b < NBW; ++b) {
            const int col = (wave * NBW + b) * 16 + col16;
            const int kb  = c * 32 + hi * 16;
            v16h bb;
            #pragma unroll
            for (int i = 0; i < 16; ++i)
                bb[i] = (_Float16)W_hh[(kb + i) * HID + col];
            b_hh[c][b] = bb;
        }
    }

    // ---- W_ph as B-operand, classes padded 10 -> 16 (used by wave 0) ----
    v16h b_ph[4];
    #pragma unroll
    for (int c = 0; c < 4; ++c) {
        const int kb = c * 32 + hi * 16;
        v16h bb;
        #pragma unroll
        for (int i = 0; i < 16; ++i)
            bb[i] = (col16 < CLASSES) ? (_Float16)W_ph[(kb + i) * CLASSES + col16]
                                      : (_Float16)0.0f;
        b_ph[c] = bb;
    }

    // ---- loop-invariant scalars in the C/D lane layout ----
    float whx[NBW], bh[NBW];
    #pragma unroll
    for (int b = 0; b < NBW; ++b) {
        const int n = (wave * NBW + b) * 16 + col16;
        whx[b] = W_hx[n];
        bh[b]  = bias_h[n];
    }

    const float* xrow = x + (size_t)(r0 + col16) * SEQ;

    __syncthreads();

    // ================= time loop (sequential dependency) =================
    for (int t = 0; t < SEQ; ++t) {
        // lanes 0-15 hold x[r0+m][t] for m = lane&15
        const float xv = xrow[t];

        // broadcast x_m into the C layout: VGPR v needs row m = v + 8*hi
        float xm[8];
        #pragma unroll
        for (int v = 0; v < 8; ++v) {
            const int srcLane = v + hi * 8;
            xm[v] = __int_as_float(
                __builtin_amdgcn_ds_bpermute(srcLane << 2, __float_as_int(xv)));
        }

        // C init = bias_h + x_t * W_hx (outer product folded into accumulator)
        v8f acc[NBW];
        #pragma unroll
        for (int b = 0; b < NBW; ++b) {
            v8f ci;
            #pragma unroll
            for (int v = 0; v < 8; ++v)
                ci[v] = fmaf(xm[v], whx[b], bh[b]);
            acc[b] = ci;
        }

        // A chunks from LDS (contiguous k order, same as B) + WMMAs for this wave's N-blocks
        const _Float16* hsrc = hbuf[t & 1];
        #pragma unroll
        for (int c = 0; c < 4; ++c) {
            const int ka = c * 32 + hi * 16; // lanes 0-15: K=0..15; 16-31: K=16..31
            union { uint4 q[2]; v16h h; } u;
            u.q[0] = *(const uint4*)&hsrc[col16 * HSTRIDE + ka];       // ds_load_b128
            u.q[1] = *(const uint4*)&hsrc[col16 * HSTRIDE + ka + 8];   // ds_load_b128
            #pragma unroll
            for (int b = 0; b < NBW; ++b)
                acc[b] = __builtin_amdgcn_wmma_f32_16x16x32_f16(
                    false, u.h, false, b_hh[c][b], (short)0, acc[b], false, false);
        }

        // hardware tanh, convert to f16, publish h_new to the other buffer
        _Float16* hdst = hbuf[(t + 1) & 1];
        #pragma unroll
        for (int b = 0; b < NBW; ++b) {
            const int n = (wave * NBW + b) * 16 + col16;
            #pragma unroll
            for (int v = 0; v < 8; ++v) {
                const int m = v + hi * 8;
                hdst[m * HSTRIDE + n] = (_Float16)fast_tanh(acc[b][v]);
            }
        }

        __syncthreads();   // one barrier per step (double buffer)
    }

    // ================= final projection: out = h_T @ W_ph + bias_p =================
    if (wave == 0) {       // wave-uniform branch: EXEC all-1s inside
        const _Float16* hfin = hbuf[SEQ & 1];
        v8f accp;
        #pragma unroll
        for (int v = 0; v < 8; ++v)
            accp[v] = (col16 < CLASSES) ? bias_p[col16] : 0.0f;

        #pragma unroll
        for (int c = 0; c < 4; ++c) {
            const int ka = c * 32 + hi * 16;
            union { uint4 q[2]; v16h h; } u;
            u.q[0] = *(const uint4*)&hfin[col16 * HSTRIDE + ka];
            u.q[1] = *(const uint4*)&hfin[col16 * HSTRIDE + ka + 8];
            accp = __builtin_amdgcn_wmma_f32_16x16x32_f16(
                false, u.h, false, b_ph[c], (short)0, accp, false, false);
        }

        if (col16 < CLASSES) {
            #pragma unroll
            for (int v = 0; v < 8; ++v) {
                const int m = v + hi * 8;
                out[(size_t)(r0 + m) * CLASSES + col16] = accp[v];
            }
        }
    }
}

extern "C" void kernel_launch(void* const* d_in, const int* in_sizes, int n_in,
                              void* d_out, int out_size, void* d_ws, size_t ws_size,
                              hipStream_t stream) {
    (void)in_sizes; (void)n_in; (void)out_size; (void)d_ws; (void)ws_size;
    const float* x      = (const float*)d_in[0];
    const float* W_hx   = (const float*)d_in[1];
    const float* W_hh   = (const float*)d_in[2];
    const float* W_ph   = (const float*)d_in[3];
    const float* bias_h = (const float*)d_in[4];
    const float* bias_p = (const float*)d_in[5];
    float* out = (float*)d_out;

    dim3 grid(BATCH / ROWS);        // 128 workgroups
    dim3 block(NWAVES * 32);        // 4 waves cooperating per 16-row tile
    rnn_wmma_kernel<<<grid, block, 0, stream>>>(x, W_hx, W_hh, W_ph, bias_h, bias_p, out);
}
